// Transformer_31817117729478
// MI455X (gfx1250) — matrix operand order repaired and checked by reading, not compiled, not run on hardware
//
#include <hip/hip_runtime.h>

typedef __attribute__((ext_vector_type(16))) _Float16 v16h;
typedef __attribute__((ext_vector_type(8)))  float    v8f;
typedef __attribute__((ext_vector_type(4)))  int      v4i;

#define NB   8
#define TT   24
#define RT   (NB * TT)      // 192 rows per block (12 M-tiles of 16)
#define DM   40
#define FFN  400
#define NH   8
#define HDM  5

// LDS layout (bytes)
#define OFF_H    0                      // h      f32 [192][40]  = 30720
#define OFF_S    30720                  // sS     f32 [192][128] = 98304
#define OFF_CTX  129024                 // cS     f32 [192][48]  = 36864
#define OFF_AF   165888                 // aF     f16 [192][64]  = 24576
#define OFF_FF   190464                 // fF     f16 [192][128] = 49152
#define OFF_W    239616                 // wF     f16 8192 elems = 16384
#define SMEM_BYTES 256000

// ---------- async global->LDS copy (CDNA5 GLOBAL_LOAD_ASYNC_TO_LDS_B128) ----------

__device__ __forceinline__ void async_copy_b128(const float* g, float* l) {
#if defined(__has_builtin) && __has_builtin(__builtin_amdgcn_global_load_async_to_lds_b128)
  __builtin_amdgcn_global_load_async_to_lds_b128((v4i*)g, (v4i*)l, 0, 0);
#else
  unsigned lo = (unsigned)(size_t)l;  // LDS generic addr: low 32 bits = LDS offset
  asm volatile("global_load_async_to_lds_b128 %0, %1, off"
               :: "v"(lo), "v"(g) : "memory");
#endif
}

__device__ __forceinline__ void async_drain() {
  asm volatile("s_wait_asynccnt 0" ::: "memory");
}

// ---------- WMMA fragment helpers (CDNA5 documented layouts) ----------

__device__ __forceinline__ v16h frag_a(const _Float16* A, int lda) {
  // A 16x32 f16: lanes 0-15 hold K 0-7 (v0-3) & 16-23 (v4-7); lanes 16-31 hold K 8-15 & 24-31
  int lane = threadIdx.x & 31;
  int m  = lane & 15;
  int kh = lane >> 4;
  v16h f;
#pragma unroll
  for (int v = 0; v < 8; ++v) {
    int k0 = ((v & 4) << 2) + (kh << 3) + ((v & 3) << 1);
    f[2 * v]     = A[m * lda + k0];
    f[2 * v + 1] = A[m * lda + k0 + 1];
  }
  return f;
}

__device__ __forceinline__ v16h frag_b(const _Float16* B, int ldb) {
  // B 32x16 f16: lanes 0-15 hold K 0-7 & 16-23, lanes 16-31 hold K 8-15 & 24-31; lane = N
  int lane = threadIdx.x & 31;
  int n  = lane & 15;
  int kh = lane >> 4;
  v16h f;
#pragma unroll
  for (int v = 0; v < 8; ++v) {
    int k0 = ((v & 4) << 2) + (kh << 3) + ((v & 3) << 1);
    f[2 * v]     = B[k0 * ldb + n];
    f[2 * v + 1] = B[(k0 + 1) * ldb + n];
  }
  return f;
}

__device__ __forceinline__ v8f frag_c_load(const float* C, int ldc) {
  int lane = threadIdx.x & 31;
  int n  = lane & 15;
  int mo = (lane >> 4) << 3;
  v8f c;
#pragma unroll
  for (int r = 0; r < 8; ++r) c[r] = C[(mo + r) * ldc + n];
  return c;
}

__device__ __forceinline__ void frag_c_store(float* C, int ldc, v8f c) {
  int lane = threadIdx.x & 31;
  int n  = lane & 15;
  int mo = (lane >> 4) << 3;
#pragma unroll
  for (int r = 0; r < 8; ++r) C[(mo + r) * ldc + n] = c[r];
}

// One 16x16 output tile per wave per job; A,B,C in LDS.
__device__ __forceinline__ void gemm_tiles(const _Float16* A, int lda,
                                           const _Float16* B, int ldb,
                                           float* C, int ldc,
                                           int Mtiles, int Ntiles, int Ksteps,
                                           bool accum) {
  int wave = threadIdx.x >> 5;
  int jobs = Mtiles * Ntiles;
  for (int job = wave; job < jobs; job += 8) {
    int mt = job / Ntiles;
    int nt = job - mt * Ntiles;
    const _Float16* At = A + mt * 16 * lda;
    const _Float16* Bt = B + nt * 16;
    float* Ct = C + mt * 16 * ldc + nt * 16;
    v8f c = {};
    if (accum) c = frag_c_load(Ct, ldc);
    for (int ks = 0; ks < Ksteps; ++ks) {
      v16h a = frag_a(At + ks * 32, lda);
      v16h b = frag_b(Bt + ks * 32 * ldb, ldb);
      c = __builtin_amdgcn_wmma_f32_16x16x32_f16(false, a, false, b,
                                                 (short)0, c, false, false);
    }
    frag_c_store(Ct, ldc, c);
  }
}

// ---------- fused transformer kernel ----------

__global__ void __launch_bounds__(256, 1)
txf_fused_kernel(const float* __restrict__ x,    const float* __restrict__ w_in,
                 const float* __restrict__ b_in, const float* __restrict__ wqkv,
                 const float* __restrict__ bqkv, const float* __restrict__ wo,
                 const float* __restrict__ bo,   const float* __restrict__ ln1s,
                 const float* __restrict__ ln1b, const float* __restrict__ w1,
                 const float* __restrict__ b1,   const float* __restrict__ w2,
                 const float* __restrict__ b2,   const float* __restrict__ ln2s,
                 const float* __restrict__ ln2b, const float* __restrict__ wout,
                 const float* __restrict__ bout, float* __restrict__ out) {
  extern __shared__ char smem[];
  float*    hS = (float*)(smem + OFF_H);      // [RT][40]
  float*    sS = (float*)(smem + OFF_S);      // [RT][128]
  float*    cS = (float*)(smem + OFF_CTX);    // [RT][48]
  _Float16* aF = (_Float16*)(smem + OFF_AF);  // [RT][64]
  _Float16* fF = (_Float16*)(smem + OFF_FF);  // [RT][128]
  _Float16* wF = (_Float16*)(smem + OFF_W);   // 8192 f16

  const int  tid   = threadIdx.x;
  const long rowG0 = (long)blockIdx.x * RT;

  // ---- stage x tile into LDS via async DMA (no VGPR round-trip) ----
  // RT*10 f32 = 480 x b128 transfers, 16B-aligned on both sides.
  {
    const float* gx = x + rowG0 * 10;
    for (int i = tid; i < RT * 10 / 4; i += 256)
      async_copy_b128(gx + i * 4, cS + i * 4);
  }
  __builtin_prefetch(wqkv, 0, 0);
  async_drain();
  __syncthreads();

  // ---- input projection + sinusoidal PE -> hS ----
  for (int i = tid; i < RT * DM; i += 256) {
    int r = i / DM, d = i - r * DM;
    int t = r % TT;
    float acc = b_in[d];
#pragma unroll
    for (int f = 0; f < 10; ++f) acc += cS[r * 10 + f] * w_in[d * 10 + f];
    // pe: exp(two_i * -ln(10000)/D), even->sin, odd->cos
    float di  = __expf((float)(d & ~1) * -0.23025850929940457f);
    float ang = (float)t * di;
    acc += (d & 1) ? __cosf(ang) : __sinf(ang);
    hS[i] = acc;
  }
  __syncthreads();

  for (int l = 0; l < 2; ++l) {
    // ---- QKV GEMM: A = f16(h) [192][64], B = wqkv^T [64][128] ----
    for (int i = tid; i < RT * 64; i += 256) {
      int r = i >> 6, k = i & 63;
      aF[i] = (k < DM) ? (_Float16)hS[r * DM + k] : (_Float16)0.f;
    }
    for (int i = tid; i < 64 * 128; i += 256) {
      int k = i >> 7, n = i & 127;
      wF[i] = (k < DM && n < 120) ? (_Float16)wqkv[l * 4800 + n * DM + k]
                                  : (_Float16)0.f;
    }
    __syncthreads();
    gemm_tiles(aF, 64, wF, 128, sS, 128, RT / 16, 8, 2, false);
    __syncthreads();
    for (int i = tid; i < RT * 120; i += 256) {
      int r = i / 120, n = i - r * 120;
      sS[r * 128 + n] += bqkv[l * 120 + n];
    }
    __syncthreads();

    // ---- attention (VALU; K=5, T=24 -> WMMA padding would waste >80%) ----
    {
      int p = tid >> 2, sub = tid & 3;   // 64 (batch,head) pairs, 4 threads each
      int bb = p >> 3, hh = p & 7;
      int rb = bb * TT;
      for (int j = 0; j < 6; ++j) {
        int q = sub + 4 * j;
        float qv[HDM];
#pragma unroll
        for (int c = 0; c < HDM; ++c) qv[c] = sS[(rb + q) * 128 + hh * HDM + c];
        float sc[TT];
        float mx = -1e30f;
        for (int kk = 0; kk < TT; ++kk) {
          float s = 0.f;
#pragma unroll
          for (int c = 0; c < HDM; ++c)
            s += qv[c] * sS[(rb + kk) * 128 + DM + hh * HDM + c];
          s *= 0.4472135954999579f;  // 1/sqrt(5)
          sc[kk] = s;
          mx = fmaxf(mx, s);
        }
        float den = 0.f;
        for (int kk = 0; kk < TT; ++kk) {
          float e = __expf(sc[kk] - mx);
          sc[kk] = e;
          den += e;
        }
        float inv = 1.f / den;
        float cv[HDM] = {0.f, 0.f, 0.f, 0.f, 0.f};
        for (int kk = 0; kk < TT; ++kk) {
          float w = sc[kk] * inv;
#pragma unroll
          for (int c = 0; c < HDM; ++c)
            cv[c] += w * sS[(rb + kk) * 128 + 2 * DM + hh * HDM + c];
        }
#pragma unroll
        for (int c = 0; c < HDM; ++c) cS[(rb + q) * 48 + hh * HDM + c] = cv[c];
      }
    }
    __syncthreads();

    // ---- out-proj GEMM: A = f16(ctx), B = wo^T [64][48] ----
    for (int i = tid; i < RT * 64; i += 256) {
      int r = i >> 6, k = i & 63;
      aF[i] = (k < DM) ? (_Float16)cS[r * 48 + k] : (_Float16)0.f;
    }
    for (int i = tid; i < 64 * 48; i += 256) {
      int k = i / 48, n = i - k * 48;
      wF[i] = (k < DM && n < DM) ? (_Float16)wo[l * 1600 + n * DM + k]
                                 : (_Float16)0.f;
    }
    __syncthreads();
    gemm_tiles(aF, 64, wF, 48, sS, 128, RT / 16, 3, 2, false);
    __syncthreads();

    // ---- residual + LayerNorm1 -> hS ----
    if (tid < RT) {
      int r = tid;
      float mu = 0.f, m2 = 0.f;
      for (int j = 0; j < DM; ++j) {
        float y = hS[r * DM + j] + sS[r * 128 + j] + bo[l * DM + j];
        mu += y;
        m2 += y * y;
      }
      mu *= (1.f / DM);
      float var = m2 * (1.f / DM) - mu * mu;
      float rs  = rsqrtf(var + 1e-5f);
      for (int j = 0; j < DM; ++j) {
        float y = hS[r * DM + j] + sS[r * 128 + j] + bo[l * DM + j];
        hS[r * DM + j] = (y - mu) * rs * ln1s[l * DM + j] + ln1b[l * DM + j];
      }
    }
    __syncthreads();

    // ---- restage A = f16(h) for FFN ----
    for (int i = tid; i < RT * 64; i += 256) {
      int r = i >> 6, k = i & 63;
      aF[i] = (k < DM) ? (_Float16)hS[r * DM + k] : (_Float16)0.f;
    }
    __syncthreads();

    // ---- FFN streamed in 128-wide chunks; FF2 accumulates into cS ----
    for (int n0 = 0; n0 < FFN; n0 += 128) {
      int cn = (FFN - n0 < 128) ? (FFN - n0) : 128;  // 128,128,128,16
      __builtin_prefetch(w2 + l * 16000 + n0, 0, 0);
      for (int i = tid; i < 64 * 128; i += 256) {
        int k = i >> 7, n = i & 127;
        wF[i] = (k < DM && n < cn) ? (_Float16)w1[l * 16000 + (n0 + n) * DM + k]
                                   : (_Float16)0.f;
      }
      __syncthreads();
      gemm_tiles(aF, 64, wF, 128, sS, 128, RT / 16, (cn + 15) / 16, 2, false);
      __syncthreads();
      // bias + relu + cvt to f16 (zero-pad cols >= cn so FF2 K-padding is exact)
      for (int i = tid; i < RT * 128; i += 256) {
        int r = i >> 7, n = i & 127;
        float v = 0.f;
        if (n < cn) v = fmaxf(sS[i] + b1[l * FFN + n0 + n], 0.f);
        fF[i] = (_Float16)v;
      }
      for (int i = tid; i < 128 * 48; i += 256) {
        int k = i / 48, n = i - k * 48;
        wF[i] = (k < cn && n < DM) ? (_Float16)w2[l * 16000 + n * FFN + n0 + k]
                                   : (_Float16)0.f;
      }
      __syncthreads();
      gemm_tiles(fF, 128, wF, 48, cS, 48, RT / 16, 3, (cn + 31) / 32, n0 != 0);
      __syncthreads();
    }

    // ---- residual + LayerNorm2 -> hS ----
    if (tid < RT) {
      int r = tid;
      float mu = 0.f, m2 = 0.f;
      for (int j = 0; j < DM; ++j) {
        float y = hS[r * DM + j] + cS[r * 48 + j] + b2[l * DM + j];
        mu += y;
        m2 += y * y;
      }
      mu *= (1.f / DM);
      float var = m2 * (1.f / DM) - mu * mu;
      float rs  = rsqrtf(var + 1e-5f);
      for (int j = 0; j < DM; ++j) {
        float y = hS[r * DM + j] + cS[r * 48 + j] + b2[l * DM + j];
        hS[r * DM + j] = (y - mu) * rs * ln2s[l * DM + j] + ln2b[l * DM + j];
      }
    }
    __syncthreads();
  }

  // ---- output head ----
  if (tid < RT) {
    float acc = bout[0];
#pragma unroll
    for (int j = 0; j < DM; ++j) acc += hS[tid * DM + j] * wout[j];
    out[rowG0 + tid] = acc;
  }
}

extern "C" void kernel_launch(void* const* d_in, const int* in_sizes, int n_in,
                              void* d_out, int out_size, void* d_ws, size_t ws_size,
                              hipStream_t stream) {
  (void)in_sizes; (void)n_in; (void)out_size; (void)d_ws; (void)ws_size;
  const float* x    = (const float*)d_in[0];
  const float* w_in = (const float*)d_in[1];
  const float* b_in = (const float*)d_in[2];
  const float* wqkv = (const float*)d_in[3];
  const float* bqkv = (const float*)d_in[4];
  const float* wo   = (const float*)d_in[5];
  const float* bo   = (const float*)d_in[6];
  const float* ln1s = (const float*)d_in[7];
  const float* ln1b = (const float*)d_in[8];
  const float* w1   = (const float*)d_in[9];
  const float* b1   = (const float*)d_in[10];
  const float* w2   = (const float*)d_in[11];
  const float* b2   = (const float*)d_in[12];
  const float* ln2s = (const float*)d_in[13];
  const float* ln2b = (const float*)d_in[14];
  const float* wout = (const float*)d_in[15];
  const float* bout = (const float*)d_in[16];
  float* out = (float*)d_out;

  dim3 grid(16384 / NB);  // 2048 workgroups x 192 rows
  dim3 block(256);        // 8 wave32 waves
  txf_fused_kernel<<<grid, block, SMEM_BYTES, stream>>>(
      x, w_in, b_in, wqkv, bqkv, wo, bo, ln1s, ln1b,
      w1, b1, w2, b2, ln2s, ln2b, wout, bout, out);
}
